// NeuralNetwork_42528766165249
// MI455X (gfx1250) — compile-verified, matrix operand order checked by reading, not run on hardware
//
#include <hip/hip_runtime.h>

// ---------------------------------------------------------------------------
// DEQ-GRU (Broyden solver) for MI455X / gfx1250.
// Matmuls run on v_wmma_f32_16x16x32_bf16 (bf16 in, f32 accumulate).
// Activations pre-converted to bf16 so the GEMM hot loop is pure load+WMMA.
// GRU recurrence is a persistent 16-WG kernel with a grid generation barrier.
// Broyden rank-update algebra is bandwidth-bound, vectorized B128 f32.
// ---------------------------------------------------------------------------

typedef __attribute__((ext_vector_type(16))) __bf16 v16bf;
typedef __attribute__((ext_vector_type(8)))  float  v8f;
typedef __attribute__((ext_vector_type(4)))  float  v4f;
typedef __attribute__((ext_vector_type(4)))  unsigned int v4u;
typedef __attribute__((ext_vector_type(4)))  unsigned short v4us;

union Frag {
  v16bf v;
  unsigned short u[16];
  v4u q[2];
};

__device__ __forceinline__ unsigned short f32_bf16(float x) {
  unsigned int u = __float_as_uint(x);
  u += 0x7FFFu + ((u >> 16) & 1u);          // round-to-nearest-even
  return (unsigned short)(u >> 16);
}

// A/B fragment per ISA 16-bit 16x32 layout: lane L holds row (L&15),
// K runs [kb, kb+8) and [kb+16, kb+24) with kb = 8*(L>>4) + k0.
__device__ __forceinline__ v16bf load_frag_bf16(const unsigned short* __restrict__ row, int kb) {
  Frag f;
  f.q[0] = *(const v4u*)(row + kb);
  f.q[1] = *(const v4u*)(row + kb + 16);
  return f.v;
}

// ---------------------------------------------------------------------------
// GEMM: C[M,N] = A_bf16[M,K] * W_bf16[N,K]^T + bias[N]
// WG = 256 threads (8 waves). Macro tile 128(M) x 128(N); wave w owns M-tile w
// and all 8 N-tiles -> 8 WMMAs per A-fragment load.
// ---------------------------------------------------------------------------
__global__ __launch_bounds__(256) void k_gemm_bias(
    const unsigned short* __restrict__ A, const unsigned short* __restrict__ W,
    const float* __restrict__ bias, float* __restrict__ C,
    int M, int N, int K) {
  const int lane = threadIdx.x & 31;
  const int wv   = threadIdx.x >> 5;
  const int n0   = blockIdx.x * 128;
  const int m0   = blockIdx.y * 128 + wv * 16;
  const int kb   = (lane >> 4) << 3;
  const unsigned short* arow = A + (size_t)(m0 + (lane & 15)) * K;
  const unsigned short* wrow[8];
#pragma unroll
  for (int j = 0; j < 8; ++j)
    wrow[j] = W + (size_t)(n0 + j * 16 + (lane & 15)) * K;

  v8f acc[8] = {};
  for (int k0 = 0; k0 < K; k0 += 32) {
    if (k0 + 32 < K) __builtin_prefetch(arow + k0 + 32 + kb, 0, 1);
    v16bf a = load_frag_bf16(arow, k0 + kb);
#pragma unroll
    for (int j = 0; j < 8; ++j) {
      v16bf b = load_frag_bf16(wrow[j], k0 + kb);
      acc[j] = __builtin_amdgcn_wmma_f32_16x16x32_bf16(false, a, false, b,
                                                       (short)0, acc[j], false, false);
    }
  }
  // D layout: vgpr r -> M = m0 + r + 8*(lane>=16), N = n0 + j*16 + (lane&15)
#pragma unroll
  for (int j = 0; j < 8; ++j) {
    int n = n0 + j * 16 + (lane & 15);
    float bv = bias[n];
#pragma unroll
    for (int r = 0; r < 8; ++r) {
      int m = m0 + r + ((lane >> 4) << 3);
      C[(size_t)m * N + n] = acc[j][r] + bv;
    }
  }
}

// ---------------------------------------------------------------------------
// Persistent GRU scan. 16 WGs; WG g owns hidden columns [g*32, g*32+32).
// Per step: gh[64x96] = h_bf16[64x512] @ Whh_slice^T via WMMA, LDS exchange,
// fused gates, state update. Two grid barriers per step (read h / write h).
// ---------------------------------------------------------------------------
#define SCAN_WGS 16

__device__ __forceinline__ void grid_sync(unsigned* cnt, unsigned* rel, unsigned gen) {
  __syncthreads();
  if (threadIdx.x == 0) {
    __threadfence();
    unsigned prev = __hip_atomic_fetch_add(cnt, 1u, __ATOMIC_ACQ_REL,
                                           __HIP_MEMORY_SCOPE_AGENT);
    if (prev == gen * SCAN_WGS - 1u) {
      __hip_atomic_store(rel, gen, __ATOMIC_RELEASE, __HIP_MEMORY_SCOPE_AGENT);
    } else {
      while (__hip_atomic_load(rel, __ATOMIC_ACQUIRE, __HIP_MEMORY_SCOPE_AGENT) < gen)
        __builtin_amdgcn_s_sleep(1);
    }
  }
  __syncthreads();
}

__global__ __launch_bounds__(256) void k_gru_scan(
    const unsigned short* __restrict__ Whh, const float* __restrict__ bhh,
    const float* __restrict__ xp, float* __restrict__ y,
    float* __restrict__ hf, unsigned short* __restrict__ hb,
    unsigned int* bar_cnt, unsigned int* bar_rel) {
  const int g    = blockIdx.x;
  const int j0   = g * 32;
  const int lane = threadIdx.x & 31;
  const int wv   = threadIdx.x >> 5;
  const int mt   = wv & 3;        // M tile (4 x 16 rows = 64 batch)
  const int wh   = wv >> 2;       // N half (2 x 48 cols)
  const int kb   = (lane >> 4) << 3;
  __shared__ float gh[64][100];   // 64 x 96 (+pad)

  // zero hidden state chunk (h0 = 0)
  for (int i = threadIdx.x; i < 64 * 32; i += 256) {
    int m = i >> 5, j = i & 31;
    hf[m * 512 + j0 + j] = 0.f;
    hb[m * 512 + j0 + j] = 0;
  }

  const unsigned short* arow = hb + (size_t)(mt * 16 + (lane & 15)) * 512;
  const unsigned short* wrow[3];
#pragma unroll
  for (int i = 0; i < 3; ++i) {
    int ti   = wh * 3 + i;              // local N tile 0..5
    int gate = ti >> 1;                 // 0=r,1=z,2=n
    int jl   = ((ti & 1) << 4) + (lane & 15);
    wrow[i]  = Whh + (size_t)(gate * 512 + j0 + jl) * 512;
  }

  for (int t = 0; t < 128; ++t) {
    grid_sync(bar_cnt, bar_rel, 2u * t + 1u);   // all h writes of t-1 visible

    v8f acc[3] = {};
    for (int k0 = 0; k0 < 512; k0 += 32) {
      v16bf a = load_frag_bf16(arow, k0 + kb);
#pragma unroll
      for (int i = 0; i < 3; ++i) {
        v16bf b = load_frag_bf16(wrow[i], k0 + kb);
        acc[i] = __builtin_amdgcn_wmma_f32_16x16x32_bf16(false, a, false, b,
                                                         (short)0, acc[i], false, false);
      }
    }
#pragma unroll
    for (int i = 0; i < 3; ++i) {
      int c = (wh * 3 + i) * 16 + (lane & 15);
#pragma unroll
      for (int r = 0; r < 8; ++r)
        gh[mt * 16 + r + ((lane >> 4) << 3)][c] = acc[i][r];
    }

    grid_sync(bar_cnt, bar_rel, 2u * t + 2u);   // everyone done reading h

    for (int i = threadIdx.x; i < 64 * 32; i += 256) {
      int b = i >> 5, j = i & 31;
      int jj = j0 + j;
      size_t xrow = (size_t)(b * 128 + t) * 1536;
      float rh = gh[b][j]      + bhh[jj];
      float zh = gh[b][32 + j] + bhh[512 + jj];
      float nh = gh[b][64 + j] + bhh[1024 + jj];
      float ri = xp[xrow + jj];
      float zi = xp[xrow + 512 + jj];
      float ni = xp[xrow + 1024 + jj];
      float r  = 1.f / (1.f + __expf(-(ri + rh)));
      float zg = 1.f / (1.f + __expf(-(zi + zh)));
      float n  = tanhf(ni + r * nh);
      float hp = hf[b * 512 + jj];
      float hn = (1.f - zg) * n + zg * hp;
      y[(size_t)(b * 128 + t) * 512 + jj] = hn;
      hf[b * 512 + jj] = hn;
      hb[b * 512 + jj] = f32_bf16(hn);
    }
    __threadfence();
  }
}

// ---------------------------------------------------------------------------
// Streamed elementwise / reduction kernels (vectorized x4 -> B128 traffic).
// All big sizes are multiples of 1024, so no tails needed on x4 kernels.
// ---------------------------------------------------------------------------
__global__ void k_cvt_bf16x4(const float* __restrict__ s, unsigned short* __restrict__ d, int n4) {
  int i = blockIdx.x * 256 + threadIdx.x;
  if (i >= n4) return;
  v4f v = *(const v4f*)(s + (size_t)i * 4);
  v4us o;
#pragma unroll
  for (int c = 0; c < 4; ++c) o[c] = f32_bf16(v[c]);
  *(v4us*)(d + (size_t)i * 4) = o;
}
__global__ void k_addx4(const float* __restrict__ a, const float* __restrict__ b,
                        float* __restrict__ o, int n4) {
  int i = blockIdx.x * 256 + threadIdx.x;
  if (i >= n4) return;
  v4f va = *(const v4f*)(a + (size_t)i * 4), vb = *(const v4f*)(b + (size_t)i * 4);
  *(v4f*)(o + (size_t)i * 4) = va + vb;
}
__global__ void k_subx4(const float* __restrict__ a, const float* __restrict__ b,
                        float* __restrict__ o, int n4) {
  int i = blockIdx.x * 256 + threadIdx.x;
  if (i >= n4) return;
  v4f va = *(const v4f*)(a + (size_t)i * 4), vb = *(const v4f*)(b + (size_t)i * 4);
  *(v4f*)(o + (size_t)i * 4) = va - vb;
}
__global__ void k_tanh_mixx4(const float* __restrict__ y, const float* __restrict__ z0,
                             const float* __restrict__ zsub, float* __restrict__ o, int n4) {
  int i = blockIdx.x * 256 + threadIdx.x;
  if (i >= n4) return;
  v4f vy = *(const v4f*)(y + (size_t)i * 4), vz = *(const v4f*)(z0 + (size_t)i * 4);
  v4f r;
#pragma unroll
  for (int c = 0; c < 4; ++c) r[c] = tanhf(vy[c] + vz[c]);
  if (zsub) r = r - *(const v4f*)(zsub + (size_t)i * 4);
  *(v4f*)(o + (size_t)i * 4) = r;
}
__global__ void k_condcopyx4(const int* flag, const float* __restrict__ s,
                             float* __restrict__ d, int n4) {
  if (*flag == 0) return;
  int i = blockIdx.x * 256 + threadIdx.x;
  if (i >= n4) return;
  *(v4f*)(d + (size_t)i * 4) = *(const v4f*)(s + (size_t)i * 4);
}
__global__ void k_init_scalars(float* lowest, int* flag) {
  if (threadIdx.x == 0) { *lowest = 1e8f; *flag = 0; }
}

#define NORM_BLKS 512
__global__ __launch_bounds__(256) void k_norm_partial(const float* __restrict__ gx,
                                                      const float* __restrict__ x,
                                                      float* __restrict__ part, int n4) {
  __shared__ float s0[256], s1[256];
  float a = 0.f, b = 0.f;
  for (int i = blockIdx.x * 256 + threadIdx.x; i < n4; i += NORM_BLKS * 256) {
    v4f gv = *(const v4f*)(gx + (size_t)i * 4);
    v4f xv = *(const v4f*)(x + (size_t)i * 4);
#pragma unroll
    for (int c = 0; c < 4; ++c) {
      float sv = gv[c] + xv[c];
      a = fmaf(gv[c], gv[c], a); b = fmaf(sv, sv, b);
    }
  }
  s0[threadIdx.x] = a; s1[threadIdx.x] = b;
  __syncthreads();
  for (int s = 128; s > 0; s >>= 1) {
    if (threadIdx.x < s) { s0[threadIdx.x] += s0[threadIdx.x + s]; s1[threadIdx.x] += s1[threadIdx.x + s]; }
    __syncthreads();
  }
  if (threadIdx.x == 0) { part[2 * blockIdx.x] = s0[0]; part[2 * blockIdx.x + 1] = s1[0]; }
}
__global__ __launch_bounds__(256) void k_norm_final(const float* __restrict__ part,
                                                    float* lowest, int* flag) {
  __shared__ float s0[256], s1[256];
  float a = 0.f, b = 0.f;
  for (int i = threadIdx.x; i < NORM_BLKS; i += 256) { a += part[2 * i]; b += part[2 * i + 1]; }
  s0[threadIdx.x] = a; s1[threadIdx.x] = b;
  __syncthreads();
  for (int s = 128; s > 0; s >>= 1) {
    if (threadIdx.x < s) { s0[threadIdx.x] += s0[threadIdx.x + s]; s1[threadIdx.x] += s1[threadIdx.x + s]; }
    __syncthreads();
  }
  if (threadIdx.x == 0) {
    float diff = sqrtf(s0[0]) / (sqrtf(s1[0]) + 1e-9f);
    int better = diff < *lowest;
    *flag = better;
    if (better) *lowest = diff;
  }
}

// out[b*T+t] = dot(X[b,:], M[b,t,:])
__global__ __launch_bounds__(256) void k_dot_slices(const float* __restrict__ X,
                                                    const float* __restrict__ Mm,
                                                    float* __restrict__ out, int T, int nsz) {
  int b = blockIdx.x / T, t = blockIdx.x % T;
  const float* xr = X + (size_t)b * nsz;
  const float* mr = Mm + ((size_t)b * T + t) * nsz;
  __shared__ float s[256];
  float a = 0.f;
  for (int i = threadIdx.x * 4; i < nsz; i += 1024) {
    v4f xv = *(const v4f*)(xr + i), mv = *(const v4f*)(mr + i);
#pragma unroll
    for (int c = 0; c < 4; ++c) a = fmaf(xv[c], mv[c], a);
  }
  s[threadIdx.x] = a; __syncthreads();
  for (int st = 128; st > 0; st >>= 1) {
    if (threadIdx.x < st) s[threadIdx.x] += s[threadIdx.x + st];
    __syncthreads();
  }
  if (threadIdx.x == 0) out[blockIdx.x] = s[0];
}

__global__ __launch_bounds__(256) void k_denom(const float* __restrict__ VTs,
                                               const float* __restrict__ dgx,
                                               float* __restrict__ den, int it, int T, int nsz) {
  int b = blockIdx.x;
  const float* vr = VTs + ((size_t)b * T + it) * nsz;
  const float* dr = dgx + (size_t)b * nsz;
  __shared__ float s[256];
  float a = 0.f;
  for (int i = threadIdx.x * 4; i < nsz; i += 1024) {
    v4f vv = *(const v4f*)(vr + i), dv = *(const v4f*)(dr + i);
#pragma unroll
    for (int c = 0; c < 4; ++c) a = fmaf(vv[c], dv[c], a);
  }
  s[threadIdx.x] = a; __syncthreads();
  for (int st = 128; st > 0; st >>= 1) {
    if (threadIdx.x < st) s[threadIdx.x] += s[threadIdx.x + st];
    __syncthreads();
  }
  if (threadIdx.x == 0) den[b] = s[0];
}

__global__ __launch_bounds__(256) void k_fc(const float* __restrict__ zf,
                                            const float* __restrict__ W,
                                            const float* __restrict__ bias,
                                            float* __restrict__ out, int nsz) {
  int b = blockIdx.x;
  const float* zr = zf + (size_t)b * nsz;
  __shared__ float s[256];
  float a = 0.f;
  for (int i = threadIdx.x * 4; i < nsz; i += 1024) {
    v4f zv = *(const v4f*)(zr + i), wv = *(const v4f*)(W + i);
#pragma unroll
    for (int c = 0; c < 4; ++c) a = fmaf(zv[c], wv[c], a);
  }
  s[threadIdx.x] = a; __syncthreads();
  for (int st = 128; st > 0; st >>= 1) {
    if (threadIdx.x < st) s[threadIdx.x] += s[threadIdx.x + st];
    __syncthreads();
  }
  if (threadIdx.x == 0) out[b] = s[0] + bias[0];
}

// vT = -dx + sum_t xTU[b,t]*VTs[b,t,:]; nan->0; write into VTs slot `it`
__global__ void k_vT(const float* __restrict__ dx, float* __restrict__ VTs,
                     const float* __restrict__ xTU, int it, int T, int nsz, int n4) {
  int i = blockIdx.x * 256 + threadIdx.x;
  if (i >= n4) return;
  size_t e = (size_t)i * 4;
  int b = (int)(e / nsz), nn = (int)(e % nsz);
  const float* base = VTs + (size_t)b * T * nsz + nn;
  const float* cf   = xTU + b * T;
  v4f acc = -*(const v4f*)(dx + e);
  for (int t = 0; t < T; ++t) {
    v4f m = *(const v4f*)(base + (size_t)t * nsz);
    float c = cf[t];
#pragma unroll
    for (int q = 0; q < 4; ++q) acc[q] = fmaf(c, m[q], acc[q]);
  }
#pragma unroll
  for (int q = 0; q < 4; ++q) if (acc[q] != acc[q]) acc[q] = 0.f;
  *(v4f*)(VTs + ((size_t)b * T + it) * nsz + nn) = acc;
}

// u = (dx + dgx - sum_t VTgx[b,t]*Us[b,t,:]) / denom[b]; nan->0; write Us slot
__global__ void k_u(const float* __restrict__ dx, const float* __restrict__ dgx,
                    float* __restrict__ Us, const float* __restrict__ VTgx,
                    const float* __restrict__ den, int it, int T, int nsz, int n4) {
  int i = blockIdx.x * 256 + threadIdx.x;
  if (i >= n4) return;
  size_t e = (size_t)i * 4;
  int b = (int)(e / nsz), nn = (int)(e % nsz);
  const float* base = Us + (size_t)b * T * nsz + nn;
  const float* cf   = VTgx + b * T;
  v4f acc = *(const v4f*)(dx + e) + *(const v4f*)(dgx + e);
  for (int t = 0; t < T; ++t) {
    v4f m = *(const v4f*)(base + (size_t)t * nsz);
    float c = -cf[t];
#pragma unroll
    for (int q = 0; q < 4; ++q) acc[q] = fmaf(c, m[q], acc[q]);
  }
  float dv = den[b];
#pragma unroll
  for (int q = 0; q < 4; ++q) {
    acc[q] /= dv;
    if (acc[q] != acc[q]) acc[q] = 0.f;
  }
  *(v4f*)(Us + ((size_t)b * T + it) * nsz + nn) = acc;
}

// update = gxn - sum_t VTx[b,t]*Us[b,t,:]   ( = -matvec(Us,VTs,gxn) )
__global__ void k_update(const float* __restrict__ gxn, const float* __restrict__ Us,
                         const float* __restrict__ VTx, float* __restrict__ upd,
                         int T, int nsz, int n4) {
  int i = blockIdx.x * 256 + threadIdx.x;
  if (i >= n4) return;
  size_t e = (size_t)i * 4;
  int b = (int)(e / nsz), nn = (int)(e % nsz);
  const float* base = Us + (size_t)b * T * nsz + nn;
  const float* cf   = VTx + b * T;
  v4f acc = *(const v4f*)(gxn + e);
  for (int t = 0; t < T; ++t) {
    v4f m = *(const v4f*)(base + (size_t)t * nsz);
    float c = -cf[t];
#pragma unroll
    for (int q = 0; q < 4; ++q) acc[q] = fmaf(c, m[q], acc[q]);
  }
  *(v4f*)(upd + e) = acc;
}

// ---------------------------------------------------------------------------
extern "C" void kernel_launch(void* const* d_in, const int* in_sizes, int n_in,
                              void* d_out, int out_size, void* d_ws, size_t ws_size,
                              hipStream_t stream) {
  const float* x     = (const float*)d_in[0];
  const float* Wih_x = (const float*)d_in[1];
  const float* Whh_x = (const float*)d_in[2];
  const float* bih_x = (const float*)d_in[3];
  const float* bhh_x = (const float*)d_in[4];
  const float* Wih_z = (const float*)d_in[5];
  const float* Whh_z = (const float*)d_in[6];
  const float* bih_z = (const float*)d_in[7];
  const float* bhh_z = (const float*)d_in[8];
  const float* Wfc   = (const float*)d_in[9];
  const float* bfc   = (const float*)d_in[10];
  (void)in_sizes; (void)n_in; (void)out_size; (void)ws_size;

  const int Bb = 64, S = 128, Dd = 128, H = 512, N3 = 1536, T = 24;
  const int NSZ = S * H;          // 65536
  const int NT  = Bb * NSZ;       // 4194304
  const int M   = Bb * S;         // 8192

  size_t off = 0;
  auto alloc = [&](size_t bytes) -> char* {
    char* p = (char*)d_ws + off;
    off += (bytes + 255) & ~(size_t)255;
    return p;
  };
  unsigned short* wihx = (unsigned short*)alloc((size_t)N3 * Dd * 2);
  unsigned short* whhx = (unsigned short*)alloc((size_t)N3 * H * 2);
  unsigned short* wihz = (unsigned short*)alloc((size_t)N3 * H * 2);
  unsigned short* whhz = (unsigned short*)alloc((size_t)N3 * H * 2);
  unsigned short* abuf = (unsigned short*)alloc((size_t)M * H * 2);   // bf16 activations
  float* xp   = (float*)alloc((size_t)M * N3 * 4);
  float* z0   = (float*)alloc((size_t)NT * 4);
  float* ybuf = (float*)alloc((size_t)NT * 4);
  float* hf   = (float*)alloc((size_t)Bb * H * 4);
  unsigned short* hb = (unsigned short*)alloc((size_t)Bb * H * 2);
  float* xeA  = (float*)alloc((size_t)NT * 4);
  float* xeB  = (float*)alloc((size_t)NT * 4);
  float* gxA  = (float*)alloc((size_t)NT * 4);
  float* gxB  = (float*)alloc((size_t)NT * 4);
  float* upd  = (float*)alloc((size_t)NT * 4);
  float* lowx = (float*)alloc((size_t)NT * 4);
  float* dgx  = (float*)alloc((size_t)NT * 4);
  float* Us   = (float*)alloc((size_t)Bb * T * NSZ * 4);
  float* VTs  = (float*)alloc((size_t)Bb * T * NSZ * 4);
  float* xTU  = (float*)alloc((size_t)Bb * T * 4);
  float* VTgx = (float*)alloc((size_t)Bb * T * 4);
  float* VTx2 = (float*)alloc((size_t)Bb * T * 4);
  float* den  = (float*)alloc((size_t)Bb * 4);
  float* part = (float*)alloc((size_t)NORM_BLKS * 2 * 4);
  float* lowest = (float*)alloc(4);
  int*   flag   = (int*)alloc(4);
  unsigned int* bar = (unsigned int*)alloc(8);

  dim3 blk(256);
  auto EW4 = [&](int n) { return dim3((unsigned)((n / 4 + 255) / 256)); };
  dim3 ggrid(N3 / 128, M / 128);

  // weights -> bf16
  k_cvt_bf16x4<<<EW4(N3 * Dd), blk, 0, stream>>>(Wih_x, wihx, N3 * Dd / 4);
  k_cvt_bf16x4<<<EW4(N3 * H),  blk, 0, stream>>>(Whh_x, whhx, N3 * H / 4);
  k_cvt_bf16x4<<<EW4(N3 * H),  blk, 0, stream>>>(Wih_z, wihz, N3 * H / 4);
  k_cvt_bf16x4<<<EW4(N3 * H),  blk, 0, stream>>>(Whh_z, whhz, N3 * H / 4);

  auto run_gru = [&](const float* A, int K, const unsigned short* wih, const float* bih,
                     const unsigned short* whh, const float* bhh, float* yout) {
    k_cvt_bf16x4<<<EW4(M * K), blk, 0, stream>>>(A, abuf, M * K / 4);
    k_gemm_bias<<<ggrid, blk, 0, stream>>>(abuf, wih, bih, xp, M, N3, K);
    hipMemsetAsync(bar, 0, 8, stream);
    k_gru_scan<<<dim3(SCAN_WGS), blk, 0, stream>>>(whh, bhh, xp, yout, hf, hb, bar, bar + 1);
  };
  auto run_g = [&](const float* z, float* gout) {
    run_gru(z, H, wihz, bih_z, whhz, bhh_z, ybuf);
    k_tanh_mixx4<<<EW4(NT), blk, 0, stream>>>(ybuf, z0, z, gout, NT / 4);
  };

  // z0 = GRU_x(x)
  run_gru(x, Dd, wihx, bih_x, whhx, bhh_x, z0);

  // Broyden init: x_est = z0, lowest_xest = z0, gx = g(z0), update = gx
  hipMemcpyAsync(xeA,  z0, (size_t)NT * 4, hipMemcpyDeviceToDevice, stream);
  hipMemcpyAsync(lowx, z0, (size_t)NT * 4, hipMemcpyDeviceToDevice, stream);
  run_g(xeA, gxA);
  hipMemcpyAsync(upd, gxA, (size_t)NT * 4, hipMemcpyDeviceToDevice, stream);
  k_init_scalars<<<dim3(1), dim3(32), 0, stream>>>(lowest, flag);
  hipMemsetAsync(Us,  0, (size_t)Bb * T * NSZ * 4, stream);
  hipMemsetAsync(VTs, 0, (size_t)Bb * T * NSZ * 4, stream);

  float* xe = xeA; float* xn = xeB; float* gc = gxA; float* gn = gxB;
  for (int it = 0; it < T; ++it) {
    k_addx4<<<EW4(NT), blk, 0, stream>>>(xe, upd, xn, NT / 4);      // x_new
    run_g(xn, gn);                                                   // gx_new
    k_subx4<<<EW4(NT), blk, 0, stream>>>(gn, gc, dgx, NT / 4);       // delta_gx
    k_norm_partial<<<dim3(NORM_BLKS), blk, 0, stream>>>(gn, xn, part, NT / 4);
    k_norm_final<<<dim3(1), blk, 0, stream>>>(part, lowest, flag);
    k_condcopyx4<<<EW4(NT), blk, 0, stream>>>(flag, xn, lowx, NT / 4);
    // dots against OLD Us/VTs first
    k_dot_slices<<<dim3(Bb * T), blk, 0, stream>>>(upd, Us,  xTU,  T, NSZ);
    k_dot_slices<<<dim3(Bb * T), blk, 0, stream>>>(dgx, VTs, VTgx, T, NSZ);
    k_vT<<<EW4(NT), blk, 0, stream>>>(upd, VTs, xTU, it, T, NSZ, NT / 4);
    k_denom<<<dim3(Bb), blk, 0, stream>>>(VTs, dgx, den, it, T, NSZ);
    k_u<<<EW4(NT), blk, 0, stream>>>(upd, dgx, Us, VTgx, den, it, T, NSZ, NT / 4);
    // update = -matvec(newUs, newVTs, gx_new)
    k_dot_slices<<<dim3(Bb * T), blk, 0, stream>>>(gn, VTs, VTx2, T, NSZ);
    k_update<<<EW4(NT), blk, 0, stream>>>(gn, Us, VTx2, upd, T, NSZ, NT / 4);
    float* tp = xe; xe = xn; xn = tp;
    tp = gc; gc = gn; gn = tp;
  }

  // z = f(lowest_xest, z0); out = flatten(z) @ Wfc^T + bfc
  run_gru(lowx, H, wihz, bih_z, whhz, bhh_z, ybuf);
  k_tanh_mixx4<<<EW4(NT), blk, 0, stream>>>(ybuf, z0, nullptr, dgx, NT / 4);
  k_fc<<<dim3(Bb), blk, 0, stream>>>(dgx, Wfc, bfc, (float*)d_out, NSZ);
}
